// Node_EncodingBlock_21466246545835
// MI455X (gfx1250) — matrix operand order repaired and checked by reading, not compiled, not run on hardware
//
#include <hip/hip_runtime.h>
#include <hip/hip_bf16.h>

typedef __attribute__((ext_vector_type(16))) _Float16 v16h;
typedef __attribute__((ext_vector_type(8)))  float    v8f;

#define B_   4
#define N_   500
#define E_   128
#define H_   8
#define D_   16
#define MS_  16
#define FF_  512
#define ROWS (B_*N_)   /* 2000 = 125 tiles of 16 exactly */
#define EPS_ 1e-5f

/* workspace byte offsets (all 256-aligned) */
#define O_PERM 0u        /* int[4][128]                       2048 B */
#define O_VG   2048u     /* f16 VgSwz [B*H][16 chunks][32][16] 512 KB */
#define O_WO   526336u   /* f16 Wo swz [4][8][32][16]          32 KB */
#define O_W1   559104u   /* f16 W1 swz [4][32][32][16]        128 KB */
#define O_W2   690176u   /* f16 W2 swz [16][8][32][16]        128 KB */
#define O_WN   821248u   /* f16 Wn swz [8][8][32][16]          64 KB */
#define O_XC   886784u   /* f32 out_concat [2000][128]          1 MB */
#define O_Y    1910784u  /* f32 y = xc@Wo+bo [2000][128]        1 MB */
#define O_FF1  2934784u  /* f32 ff1 [2000][512]                 4 MB */
#define O_Z    7030784u  /* f32 z [2000][128]                   1 MB */
#define O_ST   8054784u  /* f32 sum1,sq1,sum2,sq2 [4*128]              */
#define O_BN   8056832u  /* f32 scale1,shift1,scale2,shift2 [4*128]    */
#define O_RK   8058880u  /* f32 u[128], w0[128]                        */

/* ---------- argsort(rand) per batch via rank counting ---------- */
__global__ void perm_kernel(const float* __restrict__ rnd, int* __restrict__ perm) {
  int b = blockIdx.x, e = threadIdx.x;
  __shared__ float r[E_];
  r[e] = rnd[b*E_ + e];
  __syncthreads();
  float mine = r[e];
  int rank = 0;
  for (int j = 0; j < E_; ++j) {
    float rj = r[j];
    rank += (rj < mine) || (rj == mine && j < e);
  }
  perm[b*E_ + rank] = e;
}

/* ---------- gather V rows into WMMA B-operand register layout ---------- */
__global__ void gatherv_kernel(const float* __restrict__ Wv, const int* __restrict__ perm,
                               _Float16* __restrict__ vg) {
  int bid = blockIdx.x;            /* ((b*8 + h)*16 + chunk) */
  int chunk = bid & 15;
  int h = (bid >> 4) & 7;
  int b = bid >> 7;
  int lane = threadIdx.x;
  int col = lane & 15, laneHi = lane >> 4;
  _Float16* dst = vg + ((size_t)bid * 32 + lane) * 16;
  #pragma unroll
  for (int idx = 0; idx < 16; ++idx) {
    int k = (idx & 7) + 8 * laneHi + 16 * (idx >> 3);   /* B layout: lane = n + 16*((k>>3)&1) */
    int m = chunk * 32 + k;
    _Float16 v = (_Float16)0.0f;
    if (m < N_) {
      int pm = perm[b*E_ + (m & (E_-1))];
      v = (_Float16)Wv[pm*(H_*D_) + h*D_ + col];
    }
    dst[idx] = v;
  }
}

/* ---------- swizzle a row-major f32 [K x nCols] weight into B layout ---------- */
__global__ void swizzlew_kernel(const float* __restrict__ src, _Float16* __restrict__ dst,
                                int nTiles, int nCols) {
  int bid = blockIdx.x;            /* kc*nTiles + nt */
  int lane = threadIdx.x;
  int nt = bid % nTiles;
  int kc = bid / nTiles;
  int n = nt * 16 + (lane & 15);
  int laneHi = lane >> 4;
  _Float16* d = dst + ((size_t)bid * 32 + lane) * 16;
  #pragma unroll
  for (int idx = 0; idx < 16; ++idx) {
    int k = kc * 32 + (idx & 7) + 8 * laneHi + 16 * (idx >> 3);
    d[idx] = (_Float16)src[(size_t)k * nCols + n];
  }
}

/* ---------- rank-1 fold of demand branch: u = Wd@Wn_bot, w0 = bd@Wn_bot + bn ---------- */
__global__ void rank1_kernel(const float* __restrict__ Wd, const float* __restrict__ bd,
                             const float* __restrict__ Wn, const float* __restrict__ bnn,
                             float* __restrict__ u, float* __restrict__ w0) {
  int t = threadIdx.x;
  float su = 0.f, s0 = 0.f;
  for (int e = 0; e < E_; ++e) {
    float wv = Wn[(size_t)(E_ + e) * E_ + t];
    su = fmaf(Wd[e], wv, su);
    s0 = fmaf(bd[e], wv, s0);
  }
  u[t]  = su;
  w0[t] = s0 + bnn[t];
}

/* ---------- fused score-MLP + softmax + P@V via WMMA ---------- */
__global__ void attn_kernel(const float* __restrict__ cost,
                            const float* __restrict__ m1w, const float* __restrict__ m1b,
                            const float* __restrict__ m2w, const float* __restrict__ m2b,
                            const _Float16* __restrict__ vg, float* __restrict__ xc) {
  int bid = blockIdx.x;            /* ((b*8 + h)*32 + nt) */
  int nt = bid & 31;
  int h = (bid >> 5) & 7;
  int b = bid >> 8;
  int lane = threadIdx.x;
  int nLocal = lane & 15, laneHi = lane >> 4;
  int n = nt * 16 + nLocal;
  int nc = n < N_ ? n : (N_ - 1);
  const float* crow = cost + ((size_t)b * N_ + nc) * N_;

  float w1c[MS_], b1c[MS_], w2c[MS_];
  #pragma unroll
  for (int s = 0; s < MS_; ++s) {
    w1c[s] = m1w[h*(2*MS_) + MS_ + s];   /* dot-term weight dead: q==0 */
    b1c[s] = m1b[h*MS_ + s];
    w2c[s] = m2w[h*MS_ + s];
  }
  float b2c = m2b[h];

  const _Float16* vbase = vg + (size_t)(b*H_ + h) * 16 * 32 * 16;

  v8f c = {};
  float rsum = 0.f;
  for (int chunk = 0; chunk < 16; ++chunk) {
    v16h bt = *(const v16h*)(vbase + ((size_t)chunk * 32 + lane) * 16);
    __builtin_prefetch(crow + chunk * 32 + 40, 0, 1);
    v16h a;
    #pragma unroll
    for (int idx = 0; idx < 16; ++idx) {
      int j = idx >> 1, hf = idx & 1;
      int k = ((j & 3) << 1) + hf + (laneHi << 3) + ((j >> 2) << 4);  /* A layout */
      int m = chunk * 32 + k;
      float e = 0.f;
      if (m < N_) {
        float cv = crow[m];
        float acc = b2c;
        #pragma unroll
        for (int s = 0; s < MS_; ++s)
          acc = fmaf(fmaxf(fmaf(cv, w1c[s], b1c[s]), 0.f), w2c[s], acc);
        e = __expf(acc);   /* |acc| <= ~6, no max-subtraction needed */
      }
      rsum += e;
      a[idx] = (_Float16)e;
    }
    c = __builtin_amdgcn_wmma_f32_16x16x32_f16(false, a, false, bt, (short)0, c, false, false);
  }

  __shared__ float sred[32];
  sred[lane] = rsum;
  __syncthreads();
  #pragma unroll
  for (int r = 0; r < 8; ++r) {
    int M = r + 8 * laneHi;
    int nOut = nt * 16 + M;
    if (nOut < N_) {
      float denom = sred[M] + sred[M + 16];   /* row split across lane halves */
      xc[((size_t)b * N_ + nOut) * E_ + h * D_ + nLocal] = c[r] / denom;
    }
  }
}

/* ---------- y = xc @ Wo + bo ; accumulate BN1 stats ---------- */
__global__ void gemm_wo_kernel(const float* __restrict__ xc, const _Float16* __restrict__ woS,
                               const float* __restrict__ bo, float* __restrict__ y,
                               float* __restrict__ sum1, float* __restrict__ sq1) {
  int mt = blockIdx.x >> 3, ntile = blockIdx.x & 7;
  int lane = threadIdx.x, nLocal = lane & 15, laneHi = lane >> 4;
  int row0 = mt * 16;
  v8f c = {};
  for (int kc = 0; kc < 4; ++kc) {
    v16h bt = *(const v16h*)(woS + ((size_t)(kc*8 + ntile)*32 + lane)*16);
    const float* arow = xc + (size_t)(row0 + nLocal) * E_ + kc * 32;
    v16h a;
    #pragma unroll
    for (int idx = 0; idx < 16; ++idx) {
      int j = idx >> 1, hf = idx & 1;
      int k = ((j & 3) << 1) + hf + (laneHi << 3) + ((j >> 2) << 4);
      a[idx] = (_Float16)arow[k];
    }
    c = __builtin_amdgcn_wmma_f32_16x16x32_f16(false, a, false, bt, (short)0, c, false, false);
  }
  int nIdx = ntile * 16 + nLocal;
  float bias = bo[nIdx];
  float s = 0.f, sq = 0.f;
  #pragma unroll
  for (int r = 0; r < 8; ++r) {
    int row = row0 + r + 8 * laneHi;
    float val = c[r] + bias;
    y[(size_t)row * E_ + nIdx] = val;
    s += val; sq = fmaf(val, val, sq);
  }
  atomicAdd(&sum1[nIdx], s);
  atomicAdd(&sq1[nIdx], sq);
}

/* ---------- finalize batchnorm: scale/shift from sum/sumsq ---------- */
__global__ void bn_fin_kernel(const float* __restrict__ sum, const float* __restrict__ sq,
                              const float* __restrict__ g, const float* __restrict__ bb,
                              float* __restrict__ scale, float* __restrict__ shift) {
  int f = threadIdx.x;
  float m = sum[f] * (1.f / ROWS);
  float v = sq[f] * (1.f / ROWS) - m * m;      /* biased variance */
  float sc = g[f] * rsqrtf(v + EPS_);
  scale[f] = sc;
  shift[f] = fmaf(-m, sc, bb[f]);
}

/* ---------- ff1 = relu(bn1(y) @ W1 + b1) ; BN affine fused into f16 convert ---------- */
__global__ void ffn1_kernel(const float* __restrict__ y, const _Float16* __restrict__ w1S,
                            const float* __restrict__ b1, const float* __restrict__ scale1,
                            const float* __restrict__ shift1, float* __restrict__ ff1) {
  int mt = blockIdx.x >> 5, ntile = blockIdx.x & 31;
  int lane = threadIdx.x, nLocal = lane & 15, laneHi = lane >> 4;
  int row0 = mt * 16;
  v8f c = {};
  for (int kc = 0; kc < 4; ++kc) {
    v16h bt = *(const v16h*)(w1S + ((size_t)(kc*32 + ntile)*32 + lane)*16);
    const float* arow = y + (size_t)(row0 + nLocal) * E_;
    v16h a;
    #pragma unroll
    for (int idx = 0; idx < 16; ++idx) {
      int j = idx >> 1, hf = idx & 1;
      int k = kc*32 + ((j & 3) << 1) + hf + (laneHi << 3) + ((j >> 2) << 4);
      a[idx] = (_Float16)fmaf(arow[k], scale1[k], shift1[k]);
    }
    c = __builtin_amdgcn_wmma_f32_16x16x32_f16(false, a, false, bt, (short)0, c, false, false);
  }
  int nIdx = ntile * 16 + nLocal;
  float bias = b1[nIdx];
  #pragma unroll
  for (int r = 0; r < 8; ++r) {
    int row = row0 + r + 8 * laneHi;
    ff1[(size_t)row * FF_ + nIdx] = fmaxf(c[r] + bias, 0.f);
  }
}

/* ---------- z = bn1(y) + ff1 @ W2 + b2 ; accumulate BN2 stats ---------- */
__global__ void ffn2_kernel(const float* __restrict__ ff1, const _Float16* __restrict__ w2S,
                            const float* __restrict__ b2, const float* __restrict__ y,
                            const float* __restrict__ scale1, const float* __restrict__ shift1,
                            float* __restrict__ z, float* __restrict__ sum2, float* __restrict__ sq2) {
  int mt = blockIdx.x >> 3, ntile = blockIdx.x & 7;
  int lane = threadIdx.x, nLocal = lane & 15, laneHi = lane >> 4;
  int row0 = mt * 16;
  v8f c = {};
  for (int kc = 0; kc < 16; ++kc) {
    v16h bt = *(const v16h*)(w2S + ((size_t)(kc*8 + ntile)*32 + lane)*16);
    const float* arow = ff1 + (size_t)(row0 + nLocal) * FF_ + kc * 32;
    v16h a;
    #pragma unroll
    for (int idx = 0; idx < 16; ++idx) {
      int j = idx >> 1, hf = idx & 1;
      int k = ((j & 3) << 1) + hf + (laneHi << 3) + ((j >> 2) << 4);
      a[idx] = (_Float16)arow[k];
    }
    c = __builtin_amdgcn_wmma_f32_16x16x32_f16(false, a, false, bt, (short)0, c, false, false);
  }
  int nIdx = ntile * 16 + nLocal;
  float bias = b2[nIdx];
  float s = 0.f, sq = 0.f;
  #pragma unroll
  for (int r = 0; r < 8; ++r) {
    int row = row0 + r + 8 * laneHi;
    float out1v = fmaf(y[(size_t)row * E_ + nIdx], scale1[nIdx], shift1[nIdx]);
    float val = c[r] + bias + out1v;
    z[(size_t)row * E_ + nIdx] = val;
    s += val; sq = fmaf(val, val, sq);
  }
  atomicAdd(&sum2[nIdx], s);
  atomicAdd(&sq2[nIdx], sq);
}

/* ---------- out = bn2(z) @ Wn_top + d*u + w0 ---------- */
__global__ void final_kernel(const float* __restrict__ z, const _Float16* __restrict__ wnS,
                             const float* __restrict__ dem,
                             const float* __restrict__ scale2, const float* __restrict__ shift2,
                             const float* __restrict__ u, const float* __restrict__ w0,
                             float* __restrict__ out) {
  int mt = blockIdx.x >> 3, ntile = blockIdx.x & 7;
  int lane = threadIdx.x, nLocal = lane & 15, laneHi = lane >> 4;
  int row0 = mt * 16;
  v8f c = {};
  for (int kc = 0; kc < 4; ++kc) {    /* only top half of Wn (K=0..127) via WMMA */
    v16h bt = *(const v16h*)(wnS + ((size_t)(kc*8 + ntile)*32 + lane)*16);
    const float* arow = z + (size_t)(row0 + nLocal) * E_;
    v16h a;
    #pragma unroll
    for (int idx = 0; idx < 16; ++idx) {
      int j = idx >> 1, hf = idx & 1;
      int k = kc*32 + ((j & 3) << 1) + hf + (laneHi << 3) + ((j >> 2) << 4);
      a[idx] = (_Float16)fmaf(arow[k], scale2[k], shift2[k]);
    }
    c = __builtin_amdgcn_wmma_f32_16x16x32_f16(false, a, false, bt, (short)0, c, false, false);
  }
  int nIdx = ntile * 16 + nLocal;
  float uu = u[nIdx], ww = w0[nIdx];
  #pragma unroll
  for (int r = 0; r < 8; ++r) {
    int row = row0 + r + 8 * laneHi;
    float d = dem[row];
    out[(size_t)row * E_ + nIdx] = c[r] + fmaf(d, uu, ww);
  }
}

extern "C" void kernel_launch(void* const* d_in, const int* in_sizes, int n_in,
                              void* d_out, int out_size, void* d_ws, size_t ws_size,
                              hipStream_t stream) {
  (void)in_sizes; (void)n_in; (void)out_size; (void)ws_size;
  const float* cost = (const float*)d_in[0];
  const float* dem  = (const float*)d_in[1];
  const float* rnd  = (const float*)d_in[2];
  /* d_in[3]=Wq, d_in[4]=Wk are dead (q == 0) */
  const float* Wv   = (const float*)d_in[5];
  const float* m1w  = (const float*)d_in[6];
  const float* m1b  = (const float*)d_in[7];
  const float* m2w  = (const float*)d_in[8];
  const float* m2b  = (const float*)d_in[9];
  const float* Wo   = (const float*)d_in[10];
  const float* bo   = (const float*)d_in[11];
  const float* W1   = (const float*)d_in[12];
  const float* b1   = (const float*)d_in[13];
  const float* W2   = (const float*)d_in[14];
  const float* b2   = (const float*)d_in[15];
  const float* bn1g = (const float*)d_in[16];
  const float* bn1b = (const float*)d_in[17];
  const float* bn2g = (const float*)d_in[18];
  const float* bn2b = (const float*)d_in[19];
  const float* Wd   = (const float*)d_in[20];
  const float* bd   = (const float*)d_in[21];
  const float* Wn   = (const float*)d_in[22];
  const float* bnn  = (const float*)d_in[23];

  char* ws = (char*)d_ws;
  int*      perm = (int*)     (ws + O_PERM);
  _Float16* vg   = (_Float16*)(ws + O_VG);
  _Float16* woS  = (_Float16*)(ws + O_WO);
  _Float16* w1S  = (_Float16*)(ws + O_W1);
  _Float16* w2S  = (_Float16*)(ws + O_W2);
  _Float16* wnS  = (_Float16*)(ws + O_WN);
  float*    xc   = (float*)   (ws + O_XC);
  float*    y    = (float*)   (ws + O_Y);
  float*    ff1  = (float*)   (ws + O_FF1);
  float*    z    = (float*)   (ws + O_Z);
  float*    st   = (float*)   (ws + O_ST);
  float*    bn   = (float*)   (ws + O_BN);
  float*    rk   = (float*)   (ws + O_RK);

  hipMemsetAsync(st, 0, 4 * E_ * sizeof(float), stream);

  perm_kernel   <<<B_,        E_, 0, stream>>>(rnd, perm);
  gatherv_kernel<<<B_*H_*16,  32, 0, stream>>>(Wv, perm, vg);
  swizzlew_kernel<<<4*8,      32, 0, stream>>>(Wo, woS, 8,  E_);
  swizzlew_kernel<<<4*32,     32, 0, stream>>>(W1, w1S, 32, FF_);
  swizzlew_kernel<<<16*8,     32, 0, stream>>>(W2, w2S, 8,  E_);
  swizzlew_kernel<<<8*8,      32, 0, stream>>>(Wn, wnS, 8,  E_);
  rank1_kernel  <<<1,         E_, 0, stream>>>(Wd, bd, Wn, bnn, rk, rk + E_);

  attn_kernel   <<<B_*H_*32,  32, 0, stream>>>(cost, m1w, m1b, m2w, m2b, vg, xc);
  gemm_wo_kernel<<<125*8,     32, 0, stream>>>(xc, woS, bo, y, st, st + E_);
  bn_fin_kernel <<<1,         E_, 0, stream>>>(st, st + E_, bn1g, bn1b, bn, bn + E_);
  ffn1_kernel   <<<125*32,    32, 0, stream>>>(y, w1S, b1, bn, bn + E_, ff1);
  ffn2_kernel   <<<125*8,     32, 0, stream>>>(ff1, w2S, b2, y, bn, bn + E_, z, st + 2*E_, st + 3*E_);
  bn_fin_kernel <<<1,         E_, 0, stream>>>(st + 2*E_, st + 3*E_, bn2g, bn2b, bn + 2*E_, bn + 3*E_);
  final_kernel  <<<125*8,     32, 0, stream>>>(z, wnS, dem, bn + 2*E_, bn + 3*E_, rk, rk + E_, (float*)d_out);
}
